// QuerySAT_50998441672904
// MI455X (gfx1250) — compile-verified, matrix operand order checked by reading, not run on hardware
//
#include <hip/hip_runtime.h>
#include <hip/hip_bf16.h>
#include <math.h>

// ---------------- problem constants (from reference) ----------------
static constexpr int cNG = 20;
static constexpr int cVPG = 1000;
static constexpr int cCPG = 4200;
static constexpr int cNVARS = cNG * cVPG;        // 20000
static constexpr int cNCLAUSES = cNG * cCPG;     // 84000
static constexpr int cNNZ = cNCLAUSES * 3;       // 252000
static constexpr int cFM = 128;
static constexpr int cQM = 32;
static constexpr int cROUNDS = 8;
static constexpr float cEPS = 1e-6f;

typedef __attribute__((ext_vector_type(16))) _Float16 v16h;
typedef __attribute__((ext_vector_type(8)))  float    v8f;

// ---------------- small device helpers ----------------
__device__ __forceinline__ float softplusf(float x) {
    float ax = fabsf(x);
    return fmaxf(x, 0.0f) + log1pf(__expf(-ax));
}
__device__ __forceinline__ float sigmoidf_(float x) {
    return 1.0f / (1.0f + __expf(-x));
}

// ---------------- weight pre-pack: fragment-native f16 layout ----------------
// dst layout: [ntile j][ktile kt][lane L (32)][half h (16)]
//   n = j*16 + (L&15)
//   k = kt*32 + (h<8 ? base+h : 16+base+(h-8)),  base = (L&16)?8:0
// K remainder zero-padded, so the GEMM hot loop has no K-boundary logic.
__global__ void k_pack_w16(const float* __restrict__ W, _Float16* __restrict__ dst,
                           int K, int N) {
    int ktiles = (K + 31) >> 5;
    long total = (long)ktiles * 512 * (N >> 4);
    long i = (long)blockIdx.x * blockDim.x + threadIdx.x;
    if (i >= total) return;
    int h = (int)(i & 15);
    int L = (int)((i >> 4) & 31);
    long blk = i >> 9;
    int kt = (int)(blk % ktiles);
    int j  = (int)(blk / ktiles);
    int base = (L & 16) ? 8 : 0;
    int k = kt * 32 + ((h < 8) ? (base + h) : (16 + base + (h - 8)));
    int n = j * 16 + (L & 15);
    float v = (k < K) ? W[(long)k * N + n] : 0.0f;
    dst[i] = (_Float16)v;
}

// ---------------- state init: _zero_state ----------------
__global__ void k_init_state(float* __restrict__ x, int rows) {
    long i = (long)blockIdx.x * blockDim.x + threadIdx.x;
    long total = (long)rows * cFM;
    if (i >= total) return;
    int j = (int)(i % cFM);
    float oh = ((j == 0) ? 1.0f : 0.0f) - 1.0f / (float)cFM;
    x[i] = oh * sqrtf((float)cFM) * 0.25f;
}

// ---------------- lit rows + degrees ----------------
__global__ void k_lit_rows_deg(const int* __restrict__ fv, const int* __restrict__ fs,
                               int* __restrict__ lit_rows, float* __restrict__ lit_deg) {
    int i = blockIdx.x * blockDim.x + threadIdx.x;
    if (i >= cNNZ) return;
    int lr = fv[i] + ((fs[i] < 0) ? cNVARS : 0);
    lit_rows[i] = lr;
    atomicAdd(&lit_deg[lr], 1.0f);
}

__global__ void k_degree_weights(const float* __restrict__ lit_deg,
                                 float* __restrict__ dw, float* __restrict__ vdw) {
    int i = blockIdx.x * blockDim.x + threadIdx.x;
    if (i >= 2 * cNVARS) return;
    dw[i] = rsqrtf(fmaxf(lit_deg[i], 1.0f));
    if (i < cNVARS) {
        vdw[i] = 4.0f * rsqrtf(fmaxf(lit_deg[i] + lit_deg[i + cNVARS], 1.0f));
    }
}

// ---------------- WMMA GEMM: C[M,N] = act(A[M,K] * Wpacked + bias) ----------------
// 128 threads = 4 waves. Block tile: 64 rows x 32 cols. Each wave: 16 rows x 32 cols
// (two 16x16 WMMA tiles sharing one A fragment). B comes fragment-packed from global.
__global__ __launch_bounds__(128) void k_gemm_f16w(
    const float* __restrict__ A, const _Float16* __restrict__ Wp,
    const float* __restrict__ bias, float* __restrict__ C,
    int M, int Kd, int N, int relu)
{
    __shared__ __align__(16) _Float16 As[64 * 32];

    const int ktiles = (Kd + 31) >> 5;
    const int row0 = blockIdx.x * 64;
    const int n0   = blockIdx.y * 32;
    const int t    = threadIdx.x;
    const int wave = t >> 5;
    const int lane = t & 31;
    const int rl   = lane & 15;
    const int base = (lane & 16) ? 8 : 0;

    // staging coords (fixed per thread)
    const int sr  = t >> 1;             // 0..63
    const int skk = (t & 1) * 16;       // 0 or 16
    int gr = row0 + sr;
    if (gr >= M) gr = M - 1;            // clamp: garbage rows are masked at store
    const float* aprow = A + (long)gr * Kd + skk;

    // fragment-packed B pointers for this block's two n-tiles
    const _Float16* bp0 = Wp + ((size_t)(2 * blockIdx.y + 0) * ktiles) * 512 + lane * 16;
    const _Float16* bp1 = Wp + ((size_t)(2 * blockIdx.y + 1) * ktiles) * 512 + lane * 16;

    v8f acc0 = {}, acc1 = {};

    for (int kt = 0; kt < ktiles; ++kt) {
        const int k0 = kt * 32;
        // ---- stage A tile: 64 rows x 32 k, f32 -> f16, vector path ----
        {
            union { _Float16 h[16]; float4 f4[2]; } tmp;
            const float* ap = aprow + k0;
            if (k0 + skk + 16 <= Kd) {
                __builtin_prefetch(ap + 32, 0, 1);      // next K tile
                const float4* ap4 = (const float4*)ap;
                float4 x0 = ap4[0], x1 = ap4[1], x2 = ap4[2], x3 = ap4[3];
                tmp.h[0]  = (_Float16)x0.x; tmp.h[1]  = (_Float16)x0.y;
                tmp.h[2]  = (_Float16)x0.z; tmp.h[3]  = (_Float16)x0.w;
                tmp.h[4]  = (_Float16)x1.x; tmp.h[5]  = (_Float16)x1.y;
                tmp.h[6]  = (_Float16)x1.z; tmp.h[7]  = (_Float16)x1.w;
                tmp.h[8]  = (_Float16)x2.x; tmp.h[9]  = (_Float16)x2.y;
                tmp.h[10] = (_Float16)x2.z; tmp.h[11] = (_Float16)x2.w;
                tmp.h[12] = (_Float16)x3.x; tmp.h[13] = (_Float16)x3.y;
                tmp.h[14] = (_Float16)x3.z; tmp.h[15] = (_Float16)x3.w;
            } else {
                // K remainder (only Kd=132 path); reads stay inside the workspace
                #pragma unroll
                for (int i = 0; i < 16; ++i) {
                    int kg = k0 + skk + i;
                    float v = ap[i];
                    tmp.h[i] = (_Float16)((kg < Kd) ? v : 0.0f);
                }
            }
            float4* dst = (float4*)&As[sr * 32 + skk];
            dst[0] = tmp.f4[0];
            dst[1] = tmp.f4[1];
        }
        __syncthreads();

        // ---- fragments: A via 2x ds_load_b128, B via 4x global_load_b128 ----
        union F16Frag { v16h v; float4 f4[2]; };
        F16Frag af, b0, b1;
        const _Float16* arow = &As[(wave * 16 + rl) * 32];
        af.f4[0] = *(const float4*)(arow + base);
        af.f4[1] = *(const float4*)(arow + 16 + base);
        const float4* w0 = (const float4*)(bp0 + (size_t)kt * 512);
        const float4* w1 = (const float4*)(bp1 + (size_t)kt * 512);
        b0.f4[0] = w0[0]; b0.f4[1] = w0[1];
        b1.f4[0] = w1[0]; b1.f4[1] = w1[1];

        acc0 = __builtin_amdgcn_wmma_f32_16x16x32_f16(false, af.v, false, b0.v,
                                                      (short)0, acc0, false, false);
        acc1 = __builtin_amdgcn_wmma_f32_16x16x32_f16(false, af.v, false, b1.v,
                                                      (short)0, acc1, false, false);
        __syncthreads();
    }

    // ---- store: C/D layout: VGPR r -> M=r (+8 for lanes 16..31), N=lane&15 ----
    int mbase = row0 + wave * 16 + ((lane & 16) ? 8 : 0);
    int nA = n0 + rl;
    int nB = n0 + 16 + rl;
    float bvA = bias[nA];
    float bvB = bias[nB];
    #pragma unroll
    for (int r = 0; r < 8; ++r) {
        int m = mbase + r;
        if (m < M) {
            float vA = acc0[r] + bvA;
            float vB = acc1[r] + bvB;
            if (relu) { vA = fmaxf(vA, 0.0f); vB = fmaxf(vB, 0.0f); }
            C[(long)m * N + nA] = vA;
            C[(long)m * N + nB] = vB;
        }
    }
}

// ---------------- concat builders ----------------
__global__ void k_concat_qin(const float* __restrict__ vars, const float* __restrict__ noise_t,
                             float* __restrict__ out) {
    long i = (long)blockIdx.x * blockDim.x + threadIdx.x;
    long total = (long)cNVARS * (cFM + 4);
    if (i >= total) return;
    int r = (int)(i / (cFM + 4));
    int c = (int)(i % (cFM + 4));
    out[i] = (c < cFM) ? vars[(long)r * cFM + c] : noise_t[(long)r * 4 + (c - cFM)];
}

__global__ void k_concat_clause_in(const float* __restrict__ cstate, const float* __restrict__ cm,
                                   const float* __restrict__ cl, float* __restrict__ out) {
    long i = (long)blockIdx.x * blockDim.x + threadIdx.x;
    long total = (long)cNCLAUSES * (cFM + 2 * cQM);
    if (i >= total) return;
    int r = (int)(i / (cFM + 2 * cQM));
    int c = (int)(i % (cFM + 2 * cQM));
    float v;
    if (c < cFM)              v = cstate[(long)r * cFM + c];
    else if (c < cFM + cQM)   v = cm[(long)r * cQM + (c - cFM)];
    else                      v = cl[(long)r * cQM + (c - cFM - cQM)] * 4.0f;  // clauses_loss * 4
    out[i] = v;
}

__global__ void k_concat_unit(const float* __restrict__ vgrad, const float* __restrict__ vdw,
                              const float* __restrict__ vars, const float* __restrict__ vloss,
                              const float* __restrict__ dw, float* __restrict__ out) {
    long i = (long)blockIdx.x * blockDim.x + threadIdx.x;
    const int W = cQM + cFM + 2 * cQM;  // 224
    long total = (long)cNVARS * W;
    if (i >= total) return;
    int r = (int)(i / W);
    int c = (int)(i % W);
    float v;
    if (c < cQM)                       v = vgrad[(long)r * cQM + c] * vdw[r];
    else if (c < cQM + cFM)            v = vars[(long)r * cFM + (c - cQM)];
    else if (c < cQM + cFM + cQM)      v = vloss[(long)r * cQM + (c - cQM - cFM)] * dw[r];
    else                               v = vloss[(long)(r + cNVARS) * cQM + (c - cQM - cFM - cQM)] * dw[r + cNVARS];
    out[i] = v;
}

// ---------------- clause_val forward / vjp ----------------
__global__ void k_clause_val_fwd(const float* __restrict__ q, const int* __restrict__ fv,
                                 const int* __restrict__ fs, float* __restrict__ cl) {
    long i = (long)blockIdx.x * blockDim.x + threadIdx.x;
    long total = (long)cNCLAUSES * cQM;
    if (i >= total) return;
    int c = (int)(i / cQM);
    int d = (int)(i % cQM);
    float s = 0.0f;
    #pragma unroll
    for (int k = 0; k < 3; ++k) {
        int v = fv[c * 3 + k];
        float sg = (float)fs[c * 3 + k];
        s += softplusf(q[(long)v * cQM + d] * sg);
    }
    cl[i] = __expf(-s);
}

__global__ void k_clause_val_bwd(const float* __restrict__ q, const int* __restrict__ fv,
                                 const int* __restrict__ fs, const float* __restrict__ cl,
                                 float* __restrict__ vgrad) {
    long i = (long)blockIdx.x * blockDim.x + threadIdx.x;
    long total = (long)cNNZ * cQM;
    if (i >= total) return;
    int nnz = (int)(i / cQM);
    int d   = (int)(i % cQM);
    int c   = nnz / 3;
    int v   = fv[nnz];
    float sg = (float)fs[nnz];
    float x = q[(long)v * cQM + d] * sg;
    float g = -cl[(long)c * cQM + d] * sigmoidf_(x) * sg;
    atomicAdd(&vgrad[(long)v * cQM + d], g);
}

// ---------------- clause messages (rev_degree_weight = rsqrt(3+1) = 0.5) ----------------
__global__ void k_clause_msgs(const float* __restrict__ vmsg, const int* __restrict__ lit_rows,
                              float* __restrict__ cm) {
    long i = (long)blockIdx.x * blockDim.x + threadIdx.x;
    long total = (long)cNCLAUSES * cQM;
    if (i >= total) return;
    int c = (int)(i / cQM);
    int d = (int)(i % cQM);
    float s = 0.0f;
    #pragma unroll
    for (int k = 0; k < 3; ++k) {
        int lr = lit_rows[c * 3 + k];
        int v   = (lr < cNVARS) ? lr : (lr - cNVARS);
        int off = (lr < cNVARS) ? d  : (cQM + d);
        s += vmsg[(long)v * (2 * cQM) + off];
    }
    cm[i] = s * 0.5f;
}

// ---------------- variables_loss scatter ----------------
__global__ void k_scatter_vloss(const float* __restrict__ cdata /* ld=160, cols 0..31 */,
                                const int* __restrict__ lit_rows, float* __restrict__ vloss) {
    long i = (long)blockIdx.x * blockDim.x + threadIdx.x;
    long total = (long)cNNZ * cQM;
    if (i >= total) return;
    int nnz = (int)(i / cQM);
    int d   = (int)(i % cQM);
    int c   = nnz / 3;
    atomicAdd(&vloss[(long)lit_rows[nnz] * cQM + d], cdata[(long)c * 160 + d]);
}

// ---------------- pair norm (contiguous groups) ----------------
__global__ void k_group_mean(const float* __restrict__ x, int ld, float* __restrict__ mean,
                             int rows_per_g, int D) {
    int i = blockIdx.x * blockDim.x + threadIdx.x;
    if (i >= cNG * D) return;
    int g = i / D, d = i % D;
    const float* p = x + (long)g * rows_per_g * ld + d;
    float s = 0.0f;
    for (int r = 0; r < rows_per_g; ++r) s += p[(long)r * ld];
    mean[i] = s / (float)rows_per_g;
}

__global__ void k_group_msq(const float* __restrict__ x, int ld, const float* __restrict__ mean,
                            float* __restrict__ msq, int rows_per_g, int D) {
    __shared__ float red[256];
    int g = blockIdx.x;
    float s = 0.0f;
    long total = (long)rows_per_g * D;
    for (long i = threadIdx.x; i < total; i += blockDim.x) {
        int r = (int)(i / D), d = (int)(i % D);
        float v = x[(long)(g * rows_per_g + r) * ld + d] - mean[g * D + d];
        s += v * v;
    }
    red[threadIdx.x] = s;
    __syncthreads();
    for (int o = 128; o > 0; o >>= 1) {
        if ((int)threadIdx.x < o) red[threadIdx.x] += red[threadIdx.x + o];
        __syncthreads();
    }
    if (threadIdx.x == 0) msq[g] = red[0] / (float)rows_per_g;
}

__global__ void k_pairnorm_apply(const float* __restrict__ x, int ld, const float* __restrict__ mean,
                                 const float* __restrict__ msq, float* __restrict__ state,
                                 int rows_per_g, int D) {
    long i = (long)blockIdx.x * blockDim.x + threadIdx.x;
    long total = (long)cNG * rows_per_g * D;
    if (i >= total) return;
    long row = i / D;
    int d = (int)(i % D);
    int g = (int)(row / rows_per_g);
    float v = x[row * ld + d] - mean[g * D + d];
    state[row * D + d] = v * rsqrtf(msq[g] + cEPS) * 0.25f + 0.1f * state[row * D + d];
}

// ---------------- final 128 -> 1 layer ----------------
__global__ void k_dense_out1(const float* __restrict__ h, const float* __restrict__ w,
                             const float* __restrict__ b, float* __restrict__ logits) {
    int r = blockIdx.x * blockDim.x + threadIdx.x;
    if (r >= cNVARS) return;
    const float* hp = h + (long)r * cFM;
    float s = 0.0f;
    #pragma unroll 4
    for (int d = 0; d < cFM; ++d) s += hp[d] * w[d];
    logits[r] = s + b[0];
}

// ---------------- per-round loss ----------------
__global__ void k_loss(const float* __restrict__ logits, const int* __restrict__ fv,
                       const int* __restrict__ fs, float* __restrict__ loss_accum) {
    __shared__ float red[256];
    int g = blockIdx.x;
    float s = 0.0f;
    for (int c = g * cCPG + threadIdx.x; c < (g + 1) * cCPG; c += blockDim.x) {
        float S = 0.0f;
        #pragma unroll
        for (int k = 0; k < 3; ++k) {
            int v = fv[c * 3 + k];
            float sg = (float)fs[c * 3 + k];
            S += softplusf(logits[v] * sg);
        }
        float cv = __expf(-S);
        s += cv * (-logf(1.0f - cv + 1e-8f));
    }
    red[threadIdx.x] = s;
    __syncthreads();
    for (int o = 128; o > 0; o >>= 1) {
        if ((int)threadIdx.x < o) red[threadIdx.x] += red[threadIdx.x + o];
        __syncthreads();
    }
    if (threadIdx.x == 0) atomicAdd(loss_accum, sqrtf(red[0] + 1e-6f));
}

__global__ void k_finalize(const float* __restrict__ loss_accum, float* __restrict__ out) {
    if (blockIdx.x == 0 && threadIdx.x == 0) out[cNVARS] = loss_accum[0] / (float)cROUNDS;
}

// ---------------- host ----------------
static inline unsigned ceil_div(long a, long b) { return (unsigned)((a + b - 1) / b); }

extern "C" void kernel_launch(void* const* d_in, const int* in_sizes, int n_in,
                              void* d_out, int out_size, void* d_ws, size_t ws_size,
                              hipStream_t stream) {
    (void)in_sizes; (void)n_in; (void)out_size; (void)ws_size;

    // params in setup_inputs() insertion order:
    // variables_query (W,b)x3, lit_mlp x3, clause_mlp x3, update_gate x3, variables_output x3
    const float* W[15]; const float* B[15];
    for (int l = 0; l < 15; ++l) {
        W[l] = (const float*)d_in[2 * l];
        B[l] = (const float*)d_in[2 * l + 1];
    }
    const float* noise = (const float*)d_in[30];
    const int* fv = (const int*)d_in[31];
    const int* fs = (const int*)d_in[32];

    static const int wK[15] = {132, 64, 64, 128, 128, 128, 192, 384, 384, 224, 256, 256, 128, 128, 128};
    static const int wN[15] = { 64, 64, 32, 128, 128,  64, 384, 384, 160, 256, 256, 128, 128, 128,   1};

    // workspace carve
    char* p = (char*)d_ws;
    auto alloc = [&](size_t bytes) -> char* {
        char* r = p;
        p += (bytes + 255) & ~(size_t)255;
        return r;
    };
    _Float16* W16[15];
    size_t wpad[15];
    for (int l = 0; l < 15; ++l) {
        if (l == 14) { W16[l] = nullptr; wpad[l] = 0; continue; }  // 128x1 stays fp32
        size_t ktiles = (size_t)((wK[l] + 31) / 32);
        wpad[l] = ktiles * 512 * (size_t)(wN[l] / 16);             // padded element count
        W16[l] = (_Float16*)alloc(wpad[l] * sizeof(_Float16));
    }
    float* vars   = (float*)alloc((size_t)cNVARS * cFM * 4);
    float* cstate = (float*)alloc((size_t)cNCLAUSES * cFM * 4);
    float* bufA   = (float*)alloc((size_t)cNCLAUSES * 384 * 4);
    float* bufB   = (float*)alloc((size_t)cNCLAUSES * 384 * 4);
    float* query  = (float*)alloc((size_t)cNVARS * cQM * 4);
    float* cl     = (float*)alloc((size_t)cNCLAUSES * cQM * 4);
    float* vgrad  = (float*)alloc((size_t)cNVARS * cQM * 4);
    float* vmsg   = (float*)alloc((size_t)cNVARS * 2 * cQM * 4);
    float* cm     = (float*)alloc((size_t)cNCLAUSES * cQM * 4);
    float* cdata  = (float*)alloc((size_t)cNCLAUSES * 160 * 4);
    float* vloss  = (float*)alloc((size_t)2 * cNVARS * cQM * 4);
    int*   lit_rows = (int*)alloc((size_t)cNNZ * 4);
    float* lit_deg  = (float*)alloc((size_t)2 * cNVARS * 4);
    float* dw       = (float*)alloc((size_t)2 * cNVARS * 4);
    float* vdw      = (float*)alloc((size_t)cNVARS * 4);
    float* meanb    = (float*)alloc((size_t)cNG * cFM * 4);
    float* msqb     = (float*)alloc((size_t)cNG * 4);
    float* lossb    = (float*)alloc(4);
    float* logits   = (float*)d_out;   // d_out[0:20000] = logits, d_out[20000] = loss

    // weight pre-pack (fp32 -> fragment-native fp16), every launch: deterministic
    for (int l = 0; l < 14; ++l) {
        k_pack_w16<<<ceil_div((long)wpad[l], 256), 256, 0, stream>>>(W[l], W16[l], wK[l], wN[l]);
    }

    // init states + degrees
    k_init_state<<<ceil_div((long)cNVARS * cFM, 256), 256, 0, stream>>>(vars, cNVARS);
    k_init_state<<<ceil_div((long)cNCLAUSES * cFM, 256), 256, 0, stream>>>(cstate, cNCLAUSES);
    (void)hipMemsetAsync(lit_deg, 0, (size_t)2 * cNVARS * 4, stream);
    k_lit_rows_deg<<<ceil_div(cNNZ, 256), 256, 0, stream>>>(fv, fs, lit_rows, lit_deg);
    k_degree_weights<<<ceil_div(2 * cNVARS, 256), 256, 0, stream>>>(lit_deg, dw, vdw);
    (void)hipMemsetAsync(lossb, 0, 4, stream);

    auto gemm = [&](const float* A, const _Float16* Wc, const float* bias, float* C,
                    int M, int Kd, int N, int relu) {
        dim3 g((M + 63) / 64, N / 32);
        k_gemm_f16w<<<g, dim3(128), 0, stream>>>(A, Wc, bias, C, M, Kd, N, relu);
    };
    auto pairnorm = [&](const float* x, int ld, float* state, int rows_per_g) {
        k_group_mean<<<ceil_div(cNG * cFM, 256), 256, 0, stream>>>(x, ld, meanb, rows_per_g, cFM);
        k_group_msq<<<cNG, 256, 0, stream>>>(x, ld, meanb, msqb, rows_per_g, cFM);
        long tot = (long)cNG * rows_per_g * cFM;
        k_pairnorm_apply<<<ceil_div(tot, 256), 256, 0, stream>>>(x, ld, meanb, msqb, state, rows_per_g, cFM);
    };

    for (int r = 0; r < cROUNDS; ++r) {
        const float* noise_t = noise + (size_t)r * cNVARS * 4;

        // query = MLP([vars, noise_t])
        k_concat_qin<<<ceil_div((long)cNVARS * 132, 256), 256, 0, stream>>>(vars, noise_t, bufA);
        gemm(bufA, W16[0], B[0], bufB, cNVARS, 132, 64, 1);
        gemm(bufB, W16[1], B[1], bufA, cNVARS, 64, 64, 1);
        gemm(bufA, W16[2], B[2], query, cNVARS, 64, 32, 0);

        // clauses_loss + vjp -> variables_grad
        k_clause_val_fwd<<<ceil_div((long)cNCLAUSES * cQM, 256), 256, 0, stream>>>(query, fv, fs, cl);
        (void)hipMemsetAsync(vgrad, 0, (size_t)cNVARS * cQM * 4, stream);
        k_clause_val_bwd<<<ceil_div((long)cNNZ * cQM, 256), 256, 0, stream>>>(query, fv, fs, cl, vgrad);

        // var_msg = lit_mlp(vars)
        gemm(vars, W16[3], B[3], bufA, cNVARS, 128, 128, 1);
        gemm(bufA, W16[4], B[4], bufB, cNVARS, 128, 128, 1);
        gemm(bufB, W16[5], B[5], vmsg, cNVARS, 128, 64, 0);
        k_clause_msgs<<<ceil_div((long)cNCLAUSES * cQM, 256), 256, 0, stream>>>(vmsg, lit_rows, cm);

        // clause_data = clause_mlp([cstate, cm, cl*4])
        k_concat_clause_in<<<ceil_div((long)cNCLAUSES * 192, 256), 256, 0, stream>>>(cstate, cm, cl, bufA);
        gemm(bufA, W16[6], B[6], bufB, cNCLAUSES, 192, 384, 1);
        gemm(bufB, W16[7], B[7], bufA, cNCLAUSES, 384, 384, 1);
        gemm(bufA, W16[8], B[8], cdata, cNCLAUSES, 384, 160, 0);

        // clause_state = pair_norm(cdata[:,32:]) * 0.25 + 0.1 * clause_state
        pairnorm(cdata + cQM, 160, cstate, cCPG);

        // variables_loss scatter
        (void)hipMemsetAsync(vloss, 0, (size_t)2 * cNVARS * cQM * 4, stream);
        k_scatter_vloss<<<ceil_div((long)cNNZ * cQM, 256), 256, 0, stream>>>(cdata, lit_rows, vloss);

        // variables update
        k_concat_unit<<<ceil_div((long)cNVARS * 224, 256), 256, 0, stream>>>(vgrad, vdw, vars, vloss, dw, bufA);
        gemm(bufA, W16[9],  B[9],  bufB, cNVARS, 224, 256, 1);
        gemm(bufB, W16[10], B[10], bufA, cNVARS, 256, 256, 1);
        gemm(bufA, W16[11], B[11], bufB, cNVARS, 256, 128, 0);
        pairnorm(bufB, 128, vars, cVPG);

        // logits = variables_output(vars)
        gemm(vars, W16[12], B[12], bufA, cNVARS, 128, 128, 1);
        gemm(bufA, W16[13], B[13], bufB, cNVARS, 128, 128, 1);
        k_dense_out1<<<ceil_div(cNVARS, 256), 256, 0, stream>>>(bufB, W[14], B[14], logits);

        // per-round loss
        k_loss<<<cNG, 256, 0, stream>>>(logits, fv, fs, lossb);
    }

    k_finalize<<<1, 1, 0, stream>>>(lossb, logits);
}
